// MultiHeadAttention_25005299597754
// MI455X (gfx1250) — matrix operand order repaired and checked by reading, not compiled, not run on hardware
//
#include <hip/hip_runtime.h>

// ---------------------------------------------------------------------------
// MI455X (gfx1250) multi-head attention forward, bf16 WMMA + flash softmax.
// D_MODEL=1024, HEADS=16, HEAD_DIM=64, B=2, L=2048, fp32 in/out.
// 32-row M-tiles per wave: every B-fragment feeds two WMMAs.
// ---------------------------------------------------------------------------

typedef __bf16 bf16;
typedef __attribute__((ext_vector_type(8)))  bf16  v8bf;
typedef __attribute__((ext_vector_type(16))) bf16  v16bf;
typedef __attribute__((ext_vector_type(8)))  float v8f;

static __device__ inline v16bf cat16(v8bf lo, v8bf hi) {
  return __builtin_shufflevector(lo, hi, 0,1,2,3,4,5,6,7,8,9,10,11,12,13,14,15);
}

// A-fragment (16x32 bf16): lane l (l16=l&15, g=l>>4) holds row l16,
// K = g*8.. in elems 0..7 and K = 16+g*8.. in elems 8..15.
// Caller passes p = &row[g*8]; reads 8 halves at p and 8 at p+16.
static __device__ inline v16bf load_a16(const bf16* p) {
  v8bf lo = *(const v8bf*)p;
  v8bf hi = *(const v8bf*)(p + 16);
  return cat16(lo, hi);
}

// B-fragment (32x16 bf16): lane l holds column l&15, K = (l>>4)*8.. in elems
// 0..7 and K = 16+(l>>4)*8.. in elems 8..15 (caller passes p = &col[(l>>4)*16]).
static __device__ inline v16bf load_b16(const bf16* p) {
  const int gg = (threadIdx.x & 31) >> 4;
  v8bf lo = *(const v8bf*)(p - gg * 8);
  v8bf hi = *(const v8bf*)(p + 16 - gg * 8);
  return cat16(lo, hi);
}

static __device__ inline v8f wmma_bf16(v16bf a, v16bf b, v8f c) {
  return __builtin_amdgcn_wmma_f32_16x16x32_bf16(
      /*neg_a=*/false, a, /*neg_b=*/false, b,
      /*c_mod=*/(short)0, c, /*reuse_a=*/false, /*reuse_b=*/false);
}

// ---------------------------------------------------------------------------
// DPP cross-lane helpers (pure VALU; all exchanges stay in a 16-lane row).
// ---------------------------------------------------------------------------
#define DPP_QUAD_XOR1 0xB1   // quad_perm(1,0,3,2)  : lane ^ 1
#define DPP_QUAD_XOR2 0x4E   // quad_perm(2,3,0,1)  : lane ^ 2
#define DPP_HALF_MIRR 0x141  // row_half_mirror     : lane ^ 7
#define DPP_ROW_MIRR  0x140  // row_mirror          : lane ^ 15

template <int CTRL>
static __device__ inline float dpp_mov(float x) {
  int i = __builtin_bit_cast(int, x);
  int r = __builtin_amdgcn_update_dpp(i, i, CTRL, 0xF, 0xF, true);
  return __builtin_bit_cast(float, r);
}

static __device__ inline float row_max16(float v) {
  v = fmaxf(v, dpp_mov<DPP_QUAD_XOR1>(v));
  v = fmaxf(v, dpp_mov<DPP_QUAD_XOR2>(v));
  v = fmaxf(v, dpp_mov<DPP_HALF_MIRR>(v));
  v = fmaxf(v, dpp_mov<DPP_ROW_MIRR>(v));
  return v;
}
static __device__ inline float row_sum16(float v) {
  v += dpp_mov<DPP_QUAD_XOR1>(v);
  v += dpp_mov<DPP_QUAD_XOR2>(v);
  v += dpp_mov<DPP_HALF_MIRR>(v);
  v += dpp_mov<DPP_ROW_MIRR>(v);
  return v;
}

// ---------------------------------------------------------------------------
// fp32 -> bf16 conversion, 4 elements/thread
// ---------------------------------------------------------------------------
__global__ void cvt_bf16(const float* __restrict__ s, bf16* __restrict__ d, int n) {
  int i = (blockIdx.x * blockDim.x + threadIdx.x) * 4;
  if (i + 3 < n) {
    float4 v = *(const float4*)(s + i);
    d[i + 0] = (bf16)v.x;
    d[i + 1] = (bf16)v.y;
    d[i + 2] = (bf16)v.z;
    d[i + 3] = (bf16)v.w;
  }
}

// ---------------------------------------------------------------------------
// GEMM: out[m, n] = sum_k A[m, k] * W[n, k]  (A @ W^T, einsum 'bld,ed->ble').
// M=4096, N=1024, K=1024.  One wave per 32(M) x 64(N) slab; each B-fragment
// is reused by both 16-row A-tiles.
// ROPE:   fuse rotary embedding into the epilogue (pos = m & 2047).
// QSCALE: fold the attention 1/sqrt(64) into the stored values (exact in
//         bf16: power of two; commutes with the linear RoPE rotation).
// VTRANS: store bf16 transposed as [batch][n][pos] (for V).
// ---------------------------------------------------------------------------
template <bool ROPE, bool QSCALE, bool VTRANS, typename OutT>
__global__ __launch_bounds__(128)
void gemm16(const bf16* __restrict__ A, const bf16* __restrict__ W,
            OutT* __restrict__ out) {
  constexpr int K = 1024;
  const int lane = threadIdx.x & 31;
  const int wave = threadIdx.x >> 5;
  const int gw   = blockIdx.x * 4 + wave;       // 2048 wave-slabs
  const int m0   = (gw >> 4) * 32;              // 128 M tiles of 32
  const int n0   = (gw & 15) * 64;              // 16 N slabs
  const int g    = lane >> 4;
  const int l16  = lane & 15;

  v8f acc[2][4] = {};
  const bf16* aptr = A + (size_t)(m0 + l16) * K + g * 8;   // +16*K for mt=1
  const bf16* bptr = W + (size_t)(n0 + l16) * K + g * 16;

#pragma unroll 2
  for (int k = 0; k < K; k += 32) {
    v16bf a0 = load_a16(aptr + k);
    v16bf a1 = load_a16(aptr + (size_t)16 * K + k);
#pragma unroll
    for (int i = 0; i < 4; ++i) {
      v16bf b = load_b16(bptr + (size_t)i * 16 * K + k);
      acc[0][i] = wmma_bf16(a0, b, acc[0][i]);
      acc[1][i] = wmma_bf16(a1, b, acc[1][i]);
    }
  }

#pragma unroll
  for (int mt = 0; mt < 2; ++mt) {
#pragma unroll
    for (int i = 0; i < 4; ++i) {
#pragma unroll
      for (int j = 0; j < 8; ++j) {
        int row = m0 + mt * 16 + j + g * 8;     // C layout: VGPR j, half-wave g
        int col = n0 + i * 16 + l16;
        float v = acc[mt][i][j];
        if (ROPE) {
          float other = dpp_mov<DPP_QUAD_XOR1>(v);  // even/odd partner column
          int   pos   = row & 2047;
          int   i2    = (col & 63) & ~1;            // 2*(pair index)
          float inv   = __expf(-0.14391156683f * (float)i2);  // theta^(-i2/64)
          float ang   = (float)pos * inv;
          float c = __cosf(ang), s = __sinf(ang);
          v = (col & 1) ? fmaf(other, s, v * c)     // odd:  x*c + x_even*s
                        : fmaf(-other, s, v * c);   // even: x*c - x_odd*s
        }
        if (QSCALE) v *= 0.125f;                    // 1/sqrt(HEAD_DIM)
        if constexpr (VTRANS) {
          // [b][n][pos], pos-major for contiguous K-dim loads in attention
          out[(size_t)((row >> 11) * 1024 + col) * 2048 + (row & 2047)] = (OutT)v;
        } else {
          out[(size_t)row * 1024 + col] = (OutT)v;
        }
      }
    }
  }
}

// ---------------------------------------------------------------------------
// Flash attention: one wave per (batch, head, 32-query block).
// Q: [b, pos, h*64+d] bf16 (RoPE + 1/8 scale).  K: same, RoPE only.
// Vt: [b, h*64+d, pos] bf16.  ctx out: [b, pos, h*64+d] bf16.
// K/V fragments are shared by both 16-query sub-tiles; the two sub-tiles are
// processed sequentially through softmax to keep live ranges short.
// ---------------------------------------------------------------------------
__global__ __launch_bounds__(128)
void attn16(const bf16* __restrict__ Qm, const bf16* __restrict__ Km,
            const bf16* __restrict__ Vt, bf16* __restrict__ Cm) {
  constexpr int Lseq = 2048, Dm = 1024, HD = 64;
  __shared__ __align__(16) bf16 pbuf[4][32 * 32];   // P staging, per wave

  const int lane = threadIdx.x & 31;
  const int wave = threadIdx.x >> 5;
  const int gw   = blockIdx.x * 4 + wave;           // 2048 waves total
  const int q0   = (gw & 63) * 32;                  // 64 q-blocks of 32
  const int h    = (gw >> 6) & 15;
  const int b    = gw >> 10;
  const int g    = lane >> 4;
  const int l16  = lane & 15;

  // Preload Q A-fragments: 2 query sub-tiles x (d 0..31, d 32..63)
  const bf16* qrow = Qm + (size_t)(b * Lseq + q0 + l16) * Dm + h * HD;
  v16bf aQ[2][2];
#pragma unroll
  for (int mt = 0; mt < 2; ++mt) {
    const bf16* qr = qrow + (size_t)(mt * 16) * Dm;
    aQ[mt][0] = load_a16(qr + g * 8);
    aQ[mt][1] = load_a16(qr + 32 + g * 8);
  }

  v8f acc[2][4] = {};
  float mrow[16], drow[16];
#pragma unroll
  for (int j = 0; j < 16; ++j) { mrow[j] = -3.0e38f; drow[j] = 0.0f; }

  bf16* pw = &pbuf[wave][0];

  for (int kc = 0; kc < Lseq; kc += 32) {
    // ---- 4 shared K-fragments for this 32-key chunk ----
    const bf16* kr0 = Km + (size_t)(b * Lseq + kc + l16) * Dm + h * HD + g * 16;
    const bf16* kr1 = kr0 + (size_t)16 * Dm;
    v16bf b00 = load_b16(kr0);        // keys kc..kc+15,   d 0..31
    v16bf b01 = load_b16(kr0 + 32);   // keys kc..kc+15,   d 32..63
    v16bf b10 = load_b16(kr1);        // keys kc+16..+31,  d 0..31
    v16bf b11 = load_b16(kr1 + 32);

    // ---- per query sub-tile: scores -> online softmax -> stage P ----
#pragma unroll
    for (int mt = 0; mt < 2; ++mt) {
      v8f z = {};
      v8f s0 = wmma_bf16(aQ[mt][0], b00, z);
      s0     = wmma_bf16(aQ[mt][1], b01, s0);
      v8f s1 = wmma_bf16(aQ[mt][0], b10, z);
      s1     = wmma_bf16(aQ[mt][1], b11, s1);

#pragma unroll
      for (int j = 0; j < 8; ++j) {
        float v0 = s0[j];                           // already scaled by 1/8
        float v1 = s1[j];
        int   r  = mt * 8 + j;
        float t  = row_max16(fmaxf(v0, v1));        // DPP butterfly, 16 cols
        float mn = fmaxf(mrow[r], t);
        float f  = __expf(mrow[r] - mn);
        mrow[r]  = mn;
        drow[r] *= f;
#pragma unroll
        for (int i = 0; i < 4; ++i) acc[mt][i][j] *= f;  // rescale ctx acc
        float p0 = __expf(v0 - mn);
        float p1 = __expf(v1 - mn);
        drow[r] += row_sum16(p0 + p1);              // DPP butterfly
        // stage P (bf16) to LDS, row-major [32 rows][32 keys]
        int rj = mt * 16 + j + g * 8;
        pw[rj * 32 + l16]      = (bf16)p0;
        pw[rj * 32 + 16 + l16] = (bf16)p1;
      }
    }

    asm volatile("s_wait_dscnt 0" ::: "memory");    // P stores -> A-layout loads

    // read P back as two 16x32 A-fragments
    v16bf aP0 = load_a16(pw + (l16)      * 32 + g * 8);
    v16bf aP1 = load_a16(pw + (16 + l16) * 32 + g * 8);

    // ---- ctx += P @ V_chunk: 8 WMMAs on 4 shared V-frags ----
#pragma unroll
    for (int i = 0; i < 4; ++i) {
      const bf16* vr = Vt + (size_t)(b * Dm + h * HD + i * 16 + l16) * Lseq
                          + kc + g * 16;
      v16bf bv = load_b16(vr);
      acc[0][i] = wmma_bf16(aP0, bv, acc[0][i]);
      acc[1][i] = wmma_bf16(aP1, bv, acc[1][i]);
    }
  }

  // ---- normalize and store ctx ----
#pragma unroll
  for (int mt = 0; mt < 2; ++mt) {
#pragma unroll
    for (int i = 0; i < 4; ++i) {
#pragma unroll
      for (int j = 0; j < 8; ++j) {
        int row = b * Lseq + q0 + mt * 16 + j + g * 8;
        int col = h * HD + i * 16 + l16;
        Cm[(size_t)row * Dm + col] = (bf16)(acc[mt][i][j] / drow[mt * 8 + j]);
      }
    }
  }
}

// ---------------------------------------------------------------------------
// Host launcher
// ---------------------------------------------------------------------------
extern "C" void kernel_launch(void* const* d_in, const int* in_sizes, int n_in,
                              void* d_out, int out_size, void* d_ws, size_t ws_size,
                              hipStream_t stream) {
  (void)in_sizes; (void)n_in; (void)out_size; (void)ws_size;

  const float* x  = (const float*)d_in[0];
  const float* wq = (const float*)d_in[1];
  const float* wk = (const float*)d_in[2];
  const float* wv = (const float*)d_in[3];
  const float* wo = (const float*)d_in[4];
  float*       out = (float*)d_out;

  char* ws = (char*)d_ws;                       // 48 MB used
  bf16* xb  = (bf16*)(ws);                      // [4096,1024]        8 MB
  bf16* wqb = (bf16*)(ws + (size_t)( 8u << 20));// [1024,1024]        2 MB
  bf16* wkb = (bf16*)(ws + (size_t)(10u << 20));
  bf16* wvb = (bf16*)(ws + (size_t)(12u << 20));
  bf16* wob = (bf16*)(ws + (size_t)(14u << 20));
  bf16* Qb  = (bf16*)(ws + (size_t)(16u << 20));// [4096,1024] roped+scaled 8 MB
  bf16* Kb  = (bf16*)(ws + (size_t)(24u << 20));// [4096,1024] roped        8 MB
  bf16* Vt  = (bf16*)(ws + (size_t)(32u << 20));// [2,1024,2048]            8 MB
  bf16* Cb  = (bf16*)(ws + (size_t)(40u << 20));// ctx [4096,1024]          8 MB

  // fp32 -> bf16
  cvt_bf16<<<4096, 256, 0, stream>>>(x,  xb,  4194304);
  cvt_bf16<<<1024, 256, 0, stream>>>(wq, wqb, 1048576);
  cvt_bf16<<<1024, 256, 0, stream>>>(wk, wkb, 1048576);
  cvt_bf16<<<1024, 256, 0, stream>>>(wv, wvb, 1048576);
  cvt_bf16<<<1024, 256, 0, stream>>>(wo, wob, 1048576);

  // projections (Q: RoPE + 1/8 scale; K: RoPE; V: stored transposed)
  gemm16<true,  true,  false, bf16><<<512, 128, 0, stream>>>(xb, wqb, Qb);
  gemm16<true,  false, false, bf16><<<512, 128, 0, stream>>>(xb, wkb, Kb);
  gemm16<false, false, true,  bf16><<<512, 128, 0, stream>>>(xb, wvb, Vt);

  // flash attention
  attn16<<<512, 128, 0, stream>>>(Qb, Kb, Vt, Cb);

  // output projection, fp32 out
  gemm16<false, false, false, float><<<512, 128, 0, stream>>>(Cb, wob, out);
}
